// MoEGate_30245159698720
// MI455X (gfx1250) — matrix operand order repaired and checked by reading, not compiled, not run on hardware
//
#include <hip/hip_runtime.h>
#include <hip/hip_bf16.h>

typedef __attribute__((ext_vector_type(16))) __bf16 v16bf;
typedef __attribute__((ext_vector_type(8)))  float  v8f;

#define DIM   2048
#define NEXP  64
#define TOPK  8
#define SEQ   8192
#define BSZ   4
#define NTOK  (BSZ * SEQ)
#define TPB   256            // tokens per block (8 waves x 32 tokens)
#define ALPHA 0.1f

// ---------------------------------------------------------------------------
// Prep: split gating weight into bf16 hi/lo planes; zero aux accumulators.
// ---------------------------------------------------------------------------
__global__ __launch_bounds__(256) void moe_prep_kernel(
    const float* __restrict__ w,
    __bf16* __restrict__ w_hi,
    __bf16* __restrict__ w_lo,
    float*  __restrict__ accum /* 2*BSZ*NEXP floats */) {
  int i = blockIdx.x * blockDim.x + threadIdx.x;
  if (i < 2 * BSZ * NEXP) accum[i] = 0.0f;
  if (i < NEXP * DIM) {
    float v  = w[i];
    __bf16 h = (__bf16)v;
    w_hi[i]  = h;
    w_lo[i]  = (__bf16)(v - (float)h);
  }
}

// ---------------------------------------------------------------------------
// Main: WMMA GEMM (split-bf16, 3 products ~ f32 accuracy), two 16-token A
// tiles per wave share each B fetch (halves L2 weight traffic), then
// softmax + top-8 + aux-loss accumulation.
// ---------------------------------------------------------------------------
__global__ __launch_bounds__(256) void moe_gate_kernel(
    const float*  __restrict__ x,
    const __bf16* __restrict__ w_hi,
    const __bf16* __restrict__ w_lo,
    float* __restrict__ out_idx,
    float* __restrict__ out_w,
    float* __restrict__ score_sum,   // [BSZ][NEXP]
    float* __restrict__ ce_count) {  // [BSZ][NEXP]
  __shared__ float lds_scores[8][32][NEXP];   // 64 KB: logits -> scores in place
  __shared__ float hist[NEXP];

  const int tid      = threadIdx.x;
  const int wave     = tid >> 5;
  const int lane     = tid & 31;
  const int m        = lane & 15;     // token row within tile / expert col within tile
  const int c        = lane >> 4;     // lane half selector
  const int tokBlock = blockIdx.x * TPB;
  const int tok0     = tokBlock + wave * 32;

  if (tid < NEXP) hist[tid] = 0.0f;

  const float* xrow0 = x + (size_t)(tok0 + m) * DIM;        // A tile 0
  const float* xrow1 = xrow0 + (size_t)16 * DIM;            // A tile 1

  v8f acc[2][4] = {};   // [token tile][expert tile of 16] -> 32 tok x 64 exp

  for (int kc = 0; kc < DIM / 32; ++kc) {
    // ---- A fragments: lane holds K = kA0..kA0+15
    const int kA0 = kc * 32 + c * 16;
    float av0[16] __attribute__((aligned(16)));
    float av1[16] __attribute__((aligned(16)));
    *(float4*)(av0 + 0)  = *(const float4*)(xrow0 + kA0);
    *(float4*)(av0 + 4)  = *(const float4*)(xrow0 + kA0 + 4);
    *(float4*)(av0 + 8)  = *(const float4*)(xrow0 + kA0 + 8);
    *(float4*)(av0 + 12) = *(const float4*)(xrow0 + kA0 + 12);
    *(float4*)(av1 + 0)  = *(const float4*)(xrow1 + kA0);
    *(float4*)(av1 + 4)  = *(const float4*)(xrow1 + kA0 + 4);
    *(float4*)(av1 + 8)  = *(const float4*)(xrow1 + kA0 + 8);
    *(float4*)(av1 + 12) = *(const float4*)(xrow1 + kA0 + 12);

    v16bf ahi0, alo0, ahi1, alo1;
#pragma unroll
    for (int j = 0; j < 16; ++j) {
      __bf16 h0 = (__bf16)av0[j];
      ahi0[j]   = h0;
      alo0[j]   = (__bf16)(av0[j] - (float)h0);
      __bf16 h1 = (__bf16)av1[j];
      ahi1[j]   = h1;
      alo1[j]   = (__bf16)(av1[j] - (float)h1);
    }

    // ---- B fragments: lane = expert column, 16 contiguous K values per lane
    const int kB = kc * 32 + c * 16;
#pragma unroll
    for (int t = 0; t < 4; ++t) {
      const size_t off = (size_t)(t * 16 + m) * DIM + kB;
      v16bf bhi = *(const v16bf*)(w_hi + off);
      v16bf blo = *(const v16bf*)(w_lo + off);
      // split-precision product: ah*bh + ah*bl + al*bh  (~f32 accurate)
      acc[0][t] = __builtin_amdgcn_wmma_f32_16x16x32_bf16(false, ahi0, false, bhi,
                                                          (short)0, acc[0][t], false, false);
      acc[0][t] = __builtin_amdgcn_wmma_f32_16x16x32_bf16(false, ahi0, false, blo,
                                                          (short)0, acc[0][t], false, false);
      acc[0][t] = __builtin_amdgcn_wmma_f32_16x16x32_bf16(false, alo0, false, bhi,
                                                          (short)0, acc[0][t], false, false);
      acc[1][t] = __builtin_amdgcn_wmma_f32_16x16x32_bf16(false, ahi1, false, bhi,
                                                          (short)0, acc[1][t], false, false);
      acc[1][t] = __builtin_amdgcn_wmma_f32_16x16x32_bf16(false, ahi1, false, blo,
                                                          (short)0, acc[1][t], false, false);
      acc[1][t] = __builtin_amdgcn_wmma_f32_16x16x32_bf16(false, alo1, false, bhi,
                                                          (short)0, acc[1][t], false, false);
    }
  }

  // ---- dump logits to LDS (C/D layout: M = r + 8*c, N = m, expert = t*16 + m)
#pragma unroll
  for (int u = 0; u < 2; ++u)
#pragma unroll
    for (int t = 0; t < 4; ++t)
#pragma unroll
      for (int r = 0; r < 8; ++r)
        lds_scores[wave][u * 16 + r + 8 * c][t * 16 + m] = acc[u][t][r];

  __syncthreads();

  // ---- per-token softmax + top-8 (all 256 threads, one token each)
  {
    float* L = &lds_scores[tid >> 5][tid & 31][0];
    float mx = -3.402823466e38f;
    for (int j = 0; j < NEXP; ++j) mx = fmaxf(mx, L[j]);
    float sum = 0.0f;
    for (int j = 0; j < NEXP; ++j) { float e = expf(L[j] - mx); L[j] = e; sum += e; }
    float inv = 1.0f / sum;
    for (int j = 0; j < NEXP; ++j) L[j] *= inv;   // full softmax scores

    unsigned long long chosen = 0ull;
    float tv[TOPK]; int ti[TOPK];
    float tsum = 0.0f;
    for (int p = 0; p < TOPK; ++p) {
      float best = -1.0f; int bi = 0;
      for (int j = 0; j < NEXP; ++j) {
        if ((chosen >> j) & 1ull) continue;
        float v = L[j];
        if (v > best) { best = v; bi = j; }       // strict > : lowest index on ties
      }
      chosen |= (1ull << bi);
      tv[p] = best; ti[p] = bi; tsum += best;
      atomicAdd(&hist[bi], 1.0f);
    }
    const float wnorm = 1.0f / (tsum + 1e-20f);
    const size_t tokg = (size_t)tokBlock + tid;
    for (int p = 0; p < TOPK; ++p) {
      out_idx[tokg * TOPK + p] = (float)ti[p];
      out_w[tokg * TOPK + p]   = tv[p] * wnorm;
    }
  }
  __syncthreads();

  // ---- aux-loss accumulators: one global atomic per (batch, expert) per block
  if (tid < NEXP) {
    float s = 0.0f;
    for (int r = 0; r < TPB; ++r) s += lds_scores[r >> 5][r & 31][tid];
    const int b = tokBlock / SEQ;  // TPB divides SEQ -> whole block in one batch
    atomicAdd(&score_sum[b * NEXP + tid], s);
    atomicAdd(&ce_count[b * NEXP + tid], hist[tid]);
  }
}

// ---------------------------------------------------------------------------
// Final scalar: aux_loss = alpha * mean_b sum_e ce_norm * score_mean
// ---------------------------------------------------------------------------
__global__ __launch_bounds__(256) void moe_aux_kernel(
    const float* __restrict__ score_sum,
    const float* __restrict__ ce_count,
    float* __restrict__ aux_out) {
  __shared__ float red[256];
  const int tid = threadIdx.x;                   // tid == b*NEXP + e (4*64 = 256)
  float ce = ce_count[tid] / ((float)SEQ * (float)TOPK / (float)NEXP);
  float sm = score_sum[tid] / (float)SEQ;
  red[tid] = ce * sm;
  __syncthreads();
  for (int s = 128; s > 0; s >>= 1) {
    if (tid < s) red[tid] += red[tid + s];
    __syncthreads();
  }
  if (tid == 0) *aux_out = red[0] / (float)BSZ * ALPHA;
}

// ---------------------------------------------------------------------------
extern "C" void kernel_launch(void* const* d_in, const int* in_sizes, int n_in,
                              void* d_out, int out_size, void* d_ws, size_t ws_size,
                              hipStream_t stream) {
  const float* x = (const float*)d_in[0];   // [4, 8192, 2048] f32
  const float* w = (const float*)d_in[1];   // [64, 2048] f32

  float* out     = (float*)d_out;
  float* out_idx = out;                                 // [NTOK, 8]
  float* out_w   = out + (size_t)NTOK * TOPK;           // [NTOK, 8]
  float* aux_out = out + (size_t)2 * NTOK * TOPK;       // scalar

  // workspace layout: [score_sum 256f | ce_count 256f | pad to 2KB | w_hi | w_lo]
  char*   ws        = (char*)d_ws;
  float*  score_sum = (float*)ws;
  float*  ce_count  = score_sum + BSZ * NEXP;
  __bf16* w_hi      = (__bf16*)(ws + 2048);
  __bf16* w_lo      = (__bf16*)(ws + 2048 + (size_t)NEXP * DIM * sizeof(__bf16));

  moe_prep_kernel<<<(NEXP * DIM + 255) / 256, 256, 0, stream>>>(w, w_hi, w_lo, score_sum);
  moe_gate_kernel<<<NTOK / TPB, 256, 0, stream>>>(x, w_hi, w_lo, out_idx, out_w,
                                                  score_sum, ce_count);
  moe_aux_kernel<<<1, 256, 0, stream>>>(score_sum, ce_count, aux_out);
}
